// DiscreteVAE_25125558681957
// MI455X (gfx1250) — compile-verified, matrix operand order checked by reading, not run on hardware
//
#include <hip/hip_runtime.h>
#include <hip/hip_bf16.h>

typedef __attribute__((ext_vector_type(16))) __bf16 v16bf;
typedef __attribute__((ext_vector_type(8)))  float  v8f;
typedef unsigned short u16;

union fragU { v16bf bf; unsigned u[8]; };

// ---- bf16 helpers (round-to-nearest-even) ----
__device__ __forceinline__ u16 f2bf(float x) {
    unsigned u = __float_as_uint(x);
    unsigned r = u + 0x7FFFu + ((u >> 16) & 1u);
    return (u16)(r >> 16);
}
__device__ __forceinline__ float bf2f(u16 h) {
    return __uint_as_float(((unsigned)h) << 16);
}

// Build a 16x32 (A) or 32x16 (B) bf16 WMMA fragment from a row-major
// [rows x ldk] u16 plane. ISA layout: lane L -> row/col = L%16, K-half by L/16;
// VGPR v holds K = 2*(v%4) + 16*(v/4) + 8*(L/16) (+0,+1 packed).
__device__ __forceinline__ v16bf load_frag(const u16* plane, int ldk,
                                           int rowBase, int kbase) {
    const int lane = threadIdx.x & 31;
    const int half = lane >> 4;
    const int row  = rowBase + (lane & 15);
    fragU f;
#pragma unroll
    for (int v = 0; v < 8; ++v) {
        int k0 = kbase + 2 * (v & 3) + 16 * (v >> 2) + 8 * half;
        f.u[v] = *(const unsigned*)(plane + row * ldk + k0);
    }
    return f.bf;
}

__device__ __forceinline__ v8f wmma_bf16(v16bf a, v16bf b, v8f c) {
    return __builtin_amdgcn_wmma_f32_16x16x32_bf16(false, a, false, b,
                                                   (short)0, c, false, false);
}

// ---------------- prep kernels ----------------
__global__ void k_zero(float* out) { out[0] = 0.0f; }

__global__ void k_cvt(const float* __restrict__ src, u16* __restrict__ hi,
                      u16* __restrict__ lo, int n) {
    int i = blockIdx.x * 256 + threadIdx.x;
    if (i < n) {
        float x = src[i];
        u16 h = f2bf(x);
        hi[i] = h;
        lo[i] = f2bf(x - bf2f(h));
    }
}

// half_norm[n] = 0.5 * sum_c cb[n][c]^2   (one wave per code)
__global__ void k_norm(const float* __restrict__ cb, float* __restrict__ hn) {
    int code = blockIdx.x * 8 + (threadIdx.x >> 5);
    int lane = threadIdx.x & 31;
    float s = 0.0f;
    for (int c = lane; c < 256; c += 32) {
        float v = cb[code * 256 + c];
        s += v * v;
    }
#pragma unroll
    for (int off = 16; off > 0; off >>= 1) s += __shfl_xor(s, off, 32);
    if (lane == 0) hn[code] = 0.5f * s;
}

// ---------------- distance + partial argmax kernel ----------------
// 512 blocks x 128 threads = 2048 waves. Wave gw covers token block
// tb = gw/8 (32 tokens, two 16-row A tiles in VGPRs) and code-tile chunk
// nc = gw%8 (64 of 512 tiles). argmin d2 == argmax (x.cb - 0.5|cb|^2).
// B fragments are software-pipelined 2 deep so the scheduler can overlap
// L2 load latency with the 6 WMMAs of the current k-block.
__global__ __launch_bounds__(128) void
k_dist(const u16* __restrict__ ph, const u16* __restrict__ pl,
       const u16* __restrict__ cbh, const u16* __restrict__ cbl,
       const float* __restrict__ hn, float* __restrict__ pval,
       int* __restrict__ pidx) {
    const int wave = threadIdx.x >> 5;
    const int lane = threadIdx.x & 31;
    const int half = lane >> 4;
    const int col  = lane & 15;
    const int gw   = blockIdx.x * 4 + wave;
    const int t0   = (gw >> 3) * 32;     // token base (32 tokens)
    const int nc   = gw & 7;             // code-tile chunk (64 tiles)

    v16bf aH0[8], aL0[8], aH1[8], aL1[8];
#pragma unroll
    for (int kb = 0; kb < 8; ++kb) {
        aH0[kb] = load_frag(ph, 256, t0,      kb * 32);
        aL0[kb] = load_frag(pl, 256, t0,      kb * 32);
        aH1[kb] = load_frag(ph, 256, t0 + 16, kb * 32);
        aL1[kb] = load_frag(pl, 256, t0 + 16, kb * 32);
    }

    float best0[8], best1[8];
    int   bidx0[8], bidx1[8];
#pragma unroll
    for (int r = 0; r < 8; ++r) {
        best0[r] = -3.0e38f; bidx0[r] = 0;
        best1[r] = -3.0e38f; bidx1[r] = 0;
    }

    const int ntEnd = nc * 64 + 64;
    for (int nt = nc * 64; nt < ntEnd; ++nt) {
        const int n0 = nt * 16;
        if (nt + 1 < ntEnd) {            // prefetch next tile's rows
            __builtin_prefetch(cbh + (n0 + 16 + col) * 256, 0, 1);
            __builtin_prefetch(cbl + (n0 + 16 + col) * 256, 0, 1);
        }
        v8f c0, c1;
#pragma unroll
        for (int r = 0; r < 8; ++r) { c0[r] = 0.0f; c1[r] = 0.0f; }

        v16bf bH[2], bL[2];
        bH[0] = load_frag(cbh, 256, n0, 0);
        bL[0] = load_frag(cbl, 256, n0, 0);
#pragma unroll
        for (int kb = 0; kb < 8; ++kb) {
            const int cur = kb & 1;
            const int nxt = cur ^ 1;
            if (kb < 7) {                // issue next k-block's loads first
                bH[nxt] = load_frag(cbh, 256, n0, (kb + 1) * 32);
                bL[nxt] = load_frag(cbl, 256, n0, (kb + 1) * 32);
            }
            c0 = wmma_bf16(aH0[kb], bH[cur], c0);
            c0 = wmma_bf16(aL0[kb], bH[cur], c0);
            c0 = wmma_bf16(aH0[kb], bL[cur], c0);
            c1 = wmma_bf16(aH1[kb], bH[cur], c1);
            c1 = wmma_bf16(aL1[kb], bH[cur], c1);
            c1 = wmma_bf16(aH1[kb], bL[cur], c1);
        }
        const int n = n0 + col;
        const float h2 = hn[n];
#pragma unroll
        for (int r = 0; r < 8; ++r) {
            float s0 = c0[r] - h2;
            float s1 = c1[r] - h2;
            if (s0 > best0[r]) { best0[r] = s0; bidx0[r] = n; }
            if (s1 > best1[r]) { best1[r] = s1; bidx1[r] = n; }
        }
    }

    // reduce over the 16 columns; lane col==0 of each half holds the winner
#pragma unroll
    for (int r = 0; r < 8; ++r) {
#pragma unroll
        for (int m = 1; m < 16; m <<= 1) {
            float ov0 = __shfl_xor(best0[r], m, 32);
            int   oi0 = __shfl_xor(bidx0[r], m, 32);
            if (ov0 > best0[r] || (ov0 == best0[r] && oi0 < bidx0[r])) {
                best0[r] = ov0; bidx0[r] = oi0;
            }
            float ov1 = __shfl_xor(best1[r], m, 32);
            int   oi1 = __shfl_xor(bidx1[r], m, 32);
            if (ov1 > best1[r] || (ov1 == best1[r] && oi1 < bidx1[r])) {
                best1[r] = ov1; bidx1[r] = oi1;
            }
        }
        if (col == 0) {
            int m0 = t0 + r + 8 * half;
            pval[m0 * 8 + nc] = best0[r];
            pidx[m0 * 8 + nc] = bidx0[r];
            pval[(m0 + 16) * 8 + nc] = best1[r];
            pidx[(m0 + 16) * 8 + nc] = bidx1[r];
        }
    }
}

// final argmax over the 8 chunk candidates per token
__global__ void k_argred(const float* __restrict__ pval,
                         const int* __restrict__ pidx, int* __restrict__ tok) {
    int t = blockIdx.x * 256 + threadIdx.x;
    if (t < 8192) {
        float bv = -3.0e38f;
        int   bi = 0x7fffffff;
#pragma unroll
        for (int c = 0; c < 8; ++c) {
            float v = pval[t * 8 + c];
            int   i = pidx[t * 8 + c];
            if (v > bv || (v == bv && i < bi)) { bv = v; bi = i; }
        }
        tok[t] = bi;
    }
}

// ---------------- MLP + chamfer kernel ----------------
// 512 blocks x 128 threads, 16 tokens per block.
__global__ __launch_bounds__(128) void
k_mlp(const int* __restrict__ tok, const u16* __restrict__ cbh,
      const u16* __restrict__ cbl, const u16* __restrict__ w1h,
      const u16* __restrict__ w1l, const float* __restrict__ b1,
      const u16* __restrict__ w2h, const u16* __restrict__ w2l,
      const float* __restrict__ b2, const u16* __restrict__ w3h,
      const u16* __restrict__ w3l, const float* __restrict__ b3,
      const float* __restrict__ gt_g, float* __restrict__ out) {
    __shared__ __align__(16) char smem[61568];
    u16*   Qh   = (u16*)smem;                 // [0,16K): quantized planes
    u16*   Ql   = Qh + 4096;
    u16*   H1h  = (u16*)(smem + 16384);       // [16K,48K): h1 planes
    u16*   H1l  = H1h + 8192;
    u16*   H2h  = Qh;                         // reuse Q region for h2
    u16*   H2l  = Ql;
    float* recl = (float*)(smem + 49152);     // [48K,54K): rec f32 [16][96]
    float* gtl  = (float*)(smem + 55296);     // [54K,60K): gt  f32 [16][96]
    int*   tokl = (int*)(smem + 61440);
    float* red  = (float*)(smem + 61504);

    const int tid  = threadIdx.x;
    const int t0   = blockIdx.x * 16;
    const int wave = tid >> 5;
    const int lane = tid & 31;
    const int half = lane >> 4;
    const int col  = lane & 15;

    if (tid < 16) tokl[tid] = tok[t0 + tid];
    __syncthreads();
    for (int i = tid; i < 16 * 256; i += 128) {
        int m = i >> 8, c = i & 255;
        int tk = tokl[m];
        Qh[i] = cbh[tk * 256 + c];
        Ql[i] = cbl[tk * 256 + c];
    }
    __syncthreads();

    // ---- layer 1: [16x256] x [256x512] -> relu -> H1 planes ----
    {
        v8f acc[8];
#pragma unroll
        for (int tt = 0; tt < 8; ++tt)
#pragma unroll
            for (int r = 0; r < 8; ++r) acc[tt][r] = 0.0f;
#pragma unroll
        for (int kb = 0; kb < 8; ++kb) {
            v16bf aH = load_frag(Qh, 256, 0, kb * 32);
            v16bf aL = load_frag(Ql, 256, 0, kb * 32);
#pragma unroll
            for (int tt = 0; tt < 8; ++tt) {
                int o0 = (wave * 8 + tt) * 16;
                v16bf bH = load_frag(w1h, 256, o0, kb * 32);
                v16bf bL = load_frag(w1l, 256, o0, kb * 32);
                acc[tt] = wmma_bf16(aH, bH, acc[tt]);
                acc[tt] = wmma_bf16(aL, bH, acc[tt]);
                acc[tt] = wmma_bf16(aH, bL, acc[tt]);
            }
        }
#pragma unroll
        for (int tt = 0; tt < 8; ++tt) {
            int o = (wave * 8 + tt) * 16 + col;
            float bias = b1[o];
#pragma unroll
            for (int r = 0; r < 8; ++r) {
                int m = r + 8 * half;
                float v = fmaxf(acc[tt][r] + bias, 0.0f);
                u16 h = f2bf(v);
                H1h[m * 512 + o] = h;
                H1l[m * 512 + o] = f2bf(v - bf2f(h));
            }
        }
    }
    __syncthreads();

    // ---- layer 2: [16x512] x [512x256] -> relu -> H2 planes ----
    {
        v8f acc[4];
#pragma unroll
        for (int tt = 0; tt < 4; ++tt)
#pragma unroll
            for (int r = 0; r < 8; ++r) acc[tt][r] = 0.0f;
#pragma unroll
        for (int kb = 0; kb < 16; ++kb) {
            v16bf aH = load_frag(H1h, 512, 0, kb * 32);
            v16bf aL = load_frag(H1l, 512, 0, kb * 32);
#pragma unroll
            for (int tt = 0; tt < 4; ++tt) {
                int o0 = (wave * 4 + tt) * 16;
                v16bf bH = load_frag(w2h, 512, o0, kb * 32);
                v16bf bL = load_frag(w2l, 512, o0, kb * 32);
                acc[tt] = wmma_bf16(aH, bH, acc[tt]);
                acc[tt] = wmma_bf16(aL, bH, acc[tt]);
                acc[tt] = wmma_bf16(aH, bL, acc[tt]);
            }
        }
#pragma unroll
        for (int tt = 0; tt < 4; ++tt) {
            int o = (wave * 4 + tt) * 16 + col;
            float bias = b2[o];
#pragma unroll
            for (int r = 0; r < 8; ++r) {
                int m = r + 8 * half;
                float v = fmaxf(acc[tt][r] + bias, 0.0f);
                u16 h = f2bf(v);
                H2h[m * 256 + o] = h;
                H2l[m * 256 + o] = f2bf(v - bf2f(h));
            }
        }
    }
    __syncthreads();

    // ---- layer 3: [16x256] x [256x96] -> rec (f32 in LDS) ----
    for (int t = wave; t < 6; t += 4) {
        int o0 = t * 16;
        v8f c3;
#pragma unroll
        for (int r = 0; r < 8; ++r) c3[r] = 0.0f;
#pragma unroll
        for (int kb = 0; kb < 8; ++kb) {
            v16bf aH = load_frag(H2h, 256, 0, kb * 32);
            v16bf aL = load_frag(H2l, 256, 0, kb * 32);
            v16bf bH = load_frag(w3h, 256, o0, kb * 32);
            v16bf bL = load_frag(w3l, 256, o0, kb * 32);
            c3 = wmma_bf16(aH, bH, c3);
            c3 = wmma_bf16(aL, bH, c3);
            c3 = wmma_bf16(aH, bL, c3);
        }
        int o = o0 + col;
        float bias = b3[o];
#pragma unroll
        for (int r = 0; r < 8; ++r) {
            int m = r + 8 * half;
            recl[m * 96 + o] = c3[r] + bias;
        }
    }
    for (int i = tid; i < 16 * 96; i += 128) gtl[i] = gt_g[t0 * 96 + i];
    __syncthreads();

    // ---- chamfer (both directions) ----
    float s = 0.0f;
    for (int it = tid; it < 512; it += 128) {   // min over gt for each rec pt
        int m = it >> 5, i = it & 31;
        const float* rp = &recl[m * 96 + i * 3];
        float rx = rp[0], ry = rp[1], rz = rp[2];
        const float* gp = &gtl[m * 96];
        float dmin = 3.0e38f;
#pragma unroll
        for (int j = 0; j < 32; ++j) {
            float dx = rx - gp[j * 3 + 0];
            float dy = ry - gp[j * 3 + 1];
            float dz = rz - gp[j * 3 + 2];
            dmin = fminf(dmin, dx * dx + dy * dy + dz * dz);
        }
        s += dmin;
    }
    for (int it = tid; it < 512; it += 128) {   // min over rec for each gt pt
        int m = it >> 5, j = it & 31;
        const float* gp = &gtl[m * 96 + j * 3];
        float gx = gp[0], gy = gp[1], gz = gp[2];
        const float* rp = &recl[m * 96];
        float dmin = 3.0e38f;
#pragma unroll
        for (int i = 0; i < 32; ++i) {
            float dx = rp[i * 3 + 0] - gx;
            float dy = rp[i * 3 + 1] - gy;
            float dz = rp[i * 3 + 2] - gz;
            dmin = fminf(dmin, dx * dx + dy * dy + dz * dz);
        }
        s += dmin;
    }
#pragma unroll
    for (int off = 16; off > 0; off >>= 1) s += __shfl_xor(s, off, 32);
    if (lane == 0) red[wave] = s;
    __syncthreads();
    if (tid == 0) {
        float tot = red[0] + red[1] + red[2] + red[3];
        atomicAdd(out, tot * (1.0f / 262144.0f));  // mean over 8192*32, both dirs
    }
}

// ---------------- host launcher ----------------
extern "C" void kernel_launch(void* const* d_in, const int* in_sizes, int n_in,
                              void* d_out, int out_size, void* d_ws,
                              size_t ws_size, hipStream_t stream) {
    const float* patch = (const float*)d_in[0];   // [128,64,256]
    const float* neigh = (const float*)d_in[1];   // [128,64,32,3]
    const float* cb    = (const float*)d_in[2];   // [8192,256]
    const float* w1    = (const float*)d_in[3];   // [512,256]
    const float* b1    = (const float*)d_in[4];
    const float* w2    = (const float*)d_in[5];   // [256,512]
    const float* b2    = (const float*)d_in[6];
    const float* w3    = (const float*)d_in[7];   // [96,256]
    const float* b3    = (const float*)d_in[8];
    float* out = (float*)d_out;

    char* ws = (char*)d_ws;
    const int CB = 8192 * 256, W1 = 512 * 256, W2 = 256 * 512, W3 = 96 * 256;
    u16*   cbh  = (u16*)(ws + 0);
    u16*   cbl  = (u16*)(ws + 4194304);
    u16*   w1h  = (u16*)(ws + 8388608);
    u16*   w1l  = (u16*)(ws + 8650752);
    u16*   w2h  = (u16*)(ws + 8912896);
    u16*   w2l  = (u16*)(ws + 9175040);
    u16*   w3h  = (u16*)(ws + 9437184);
    u16*   w3l  = (u16*)(ws + 9486336);
    float* hn   = (float*)(ws + 9535488);
    int*   tid  = (int*)(ws + 9568256);
    u16*   ph   = (u16*)(ws + 9601024);
    u16*   pl   = (u16*)(ws + 13795328);
    float* pval = (float*)(ws + 17989632);
    int*   pidx = (int*)(ws + 18251776);

    k_zero<<<1, 1, 0, stream>>>(out);
    k_cvt<<<(CB + 255) / 256, 256, 0, stream>>>(cb, cbh, cbl, CB);
    k_cvt<<<(CB + 255) / 256, 256, 0, stream>>>(patch, ph, pl, CB);
    k_cvt<<<(W1 + 255) / 256, 256, 0, stream>>>(w1, w1h, w1l, W1);
    k_cvt<<<(W2 + 255) / 256, 256, 0, stream>>>(w2, w2h, w2l, W2);
    k_cvt<<<(W3 + 255) / 256, 256, 0, stream>>>(w3, w3h, w3l, W3);
    k_norm<<<1024, 256, 0, stream>>>(cb, hn);
    k_dist<<<512, 128, 0, stream>>>(ph, pl, cbh, cbl, hn, pval, pidx);
    k_argred<<<32, 256, 0, stream>>>(pval, pidx, tid);
    k_mlp<<<512, 128, 0, stream>>>(tid, cbh, cbl, w1h, w1l, b1, w2h, w2l, b2,
                                   w3h, w3l, b3, neigh, out);
    (void)in_sizes; (void)n_in; (void)out_size; (void)ws_size;
}
